// CausalSelfAttention_5634997092957
// MI455X (gfx1250) — matrix operand order repaired and checked by reading, not compiled, not run on hardware
//
#include <hip/hip_runtime.h>
#include <stdint.h>

// ---------------------------------------------------------------------------
// CDNA5 (gfx1250) causal self-attention: bf16 WMMA everywhere, f32 accumulate.
// Pipeline: QKV gemm -> RoPE -> flash attention -> output projection.
// ---------------------------------------------------------------------------

typedef __bf16 bf16_t;
typedef __attribute__((ext_vector_type(16))) __bf16 v16bf;
typedef __attribute__((ext_vector_type(8)))  float  v8f;

struct U2 { uint4 a, b; };

__device__ __forceinline__ bf16_t f2bf(float f) {
  unsigned u = __builtin_bit_cast(unsigned, f);
  unsigned r = u + 0x7FFFu + ((u >> 16) & 1u);   // round-to-nearest-even
  unsigned short h = (unsigned short)(r >> 16);
  return __builtin_bit_cast(bf16_t, h);
}
__device__ __forceinline__ float bf2f(bf16_t b) {
  unsigned short h = __builtin_bit_cast(unsigned short, b);
  return __builtin_bit_cast(float, ((unsigned)h) << 16);
}

__device__ __forceinline__ v8f wmma_bf16(v16bf a, v16bf b, v8f c) {
  // D = A(16x32 bf16) * B(32x16 bf16) + C(16x16 f32)
  return __builtin_amdgcn_wmma_f32_16x16x32_bf16(false, a, false, b, (short)0, c,
                                                 false, false);
}

// A fragment: 16(M) x 32(K), source tile row-major [row][k], k contiguous.
// ISA layout: lanes 0-15 hold M=lane, K in {0..7,16..23}; lanes 16-31 hold
// M=lane-16, K in {8..15,24..31}; two 16B chunks per lane.
__device__ __forceinline__ v16bf load_frag_a(const bf16_t* base, int stride, int lane) {
  int row = lane & 15;
  int kb0 = (lane & 16) ? 8 : 0;
  const bf16_t* p = base + row * stride;
  U2 t;
  t.a = *(const uint4*)(p + kb0);
  t.b = *(const uint4*)(p + kb0 + 16);
  return __builtin_bit_cast(v16bf, t);
}

// B fragment: 32(K) x 16(N), source tile stored TRANSPOSED [n][k], k contiguous.
// ISA layout: lanes 0-15 hold N=lane with K in {0..7,16..23}; lanes 16-31 hold
// N=lane-16 with K in {8..15,24..31}; two 16B chunks per lane.
__device__ __forceinline__ v16bf load_frag_b(const bf16_t* baseT, int stride, int lane) {
  int n = lane & 15;
  int kb = (lane & 16) ? 8 : 0;
  const bf16_t* p = baseT + n * stride + kb;
  U2 t;
  t.a = *(const uint4*)(p);
  t.b = *(const uint4*)(p + 16);
  return __builtin_bit_cast(v16bf, t);
}

// ---------------------------------------------------------------------------
// Kernel 1: fused QKV projection.  [8192x1024] x [1024x1024] (+bias), x3.
// Output scattered as bf16 into [B, H, T, Dh].
// ---------------------------------------------------------------------------
#define GM_AS 40   // LDS row stride (bf16 elems): 80B rows keep 16B chunk alignment
#define GM_BS 40

__global__ __launch_bounds__(256) void qkv_gemm_kernel(
    const float* __restrict__ X,
    const float* __restrict__ Wq, const float* __restrict__ Wk, const float* __restrict__ Wv,
    const float* __restrict__ bq, const float* __restrict__ bk, const float* __restrict__ bv,
    bf16_t* __restrict__ Qo, bf16_t* __restrict__ Ko, bf16_t* __restrict__ Vo)
{
  const int N = 1024, K = 1024;
  const float* W; const float* bias; bf16_t* dst;
  if (blockIdx.z == 0)      { W = Wq; bias = bq; dst = Qo; }
  else if (blockIdx.z == 1) { W = Wk; bias = bk; dst = Ko; }
  else                      { W = Wv; bias = bv; dst = Vo; }

  __shared__ __align__(16) bf16_t As[128][GM_AS];   // [m][k]
  __shared__ __align__(16) bf16_t Bt[64][GM_BS];    // [n][k] (transposed)

  const int tid  = threadIdx.x;
  const int lane = tid & 31, wave = tid >> 5;
  const int wm = wave & 3, wn = wave >> 2;
  const int m0 = blockIdx.y * 128, n0 = blockIdx.x * 64;
  const int rowBase = wm * 32, colBase = wn * 32;

  v8f acc[2][2] = {};

  for (int k0 = 0; k0 < K; k0 += 32) {
    __syncthreads();
    #pragma unroll
    for (int i = 0; i < 16; ++i) {               // 128x32 A tile, f32 -> bf16
      int idx = tid + i * 256;
      int r = idx >> 5, c = idx & 31;
      As[r][c] = f2bf(X[(size_t)(m0 + r) * K + (k0 + c)]);
    }
    #pragma unroll
    for (int i = 0; i < 8; ++i) {                // 32x64 W tile, transposed into LDS
      int idx = tid + i * 256;
      int kk = idx >> 6, n = idx & 63;
      Bt[n][kk] = f2bf(W[(size_t)(k0 + kk) * N + (n0 + n)]);
    }
    if (k0 + 32 < K)                             // speculative prefetch of next A tile
      __builtin_prefetch(&X[(size_t)(m0 + (tid >> 1)) * K + k0 + 32], 0, 0);
    __syncthreads();

    v16bf a0 = load_frag_a(&As[rowBase][0],      GM_AS, lane);
    v16bf a1 = load_frag_a(&As[rowBase + 16][0], GM_AS, lane);
    v16bf b0 = load_frag_b(&Bt[colBase][0],      GM_BS, lane);
    v16bf b1 = load_frag_b(&Bt[colBase + 16][0], GM_BS, lane);
    acc[0][0] = wmma_bf16(a0, b0, acc[0][0]);
    acc[0][1] = wmma_bf16(a0, b1, acc[0][1]);
    acc[1][0] = wmma_bf16(a1, b0, acc[1][0]);
    acc[1][1] = wmma_bf16(a1, b1, acc[1][1]);
  }

  const int ncol = lane & 15, mo = (lane & 16) ? 8 : 0;
  #pragma unroll
  for (int tm = 0; tm < 2; ++tm) {
    #pragma unroll
    for (int tn = 0; tn < 2; ++tn) {
      int n = n0 + colBase + tn * 16 + ncol;
      float bb = bias[n];
      int h = n >> 6, d = n & 63;
      #pragma unroll
      for (int r = 0; r < 8; ++r) {
        int m = m0 + rowBase + tm * 16 + r + mo;  // C/D layout: VGPR r -> M, lane -> N
        int b = m >> 11, t = m & 2047;
        dst[(((size_t)b * 16 + h) * 2048 + t) * 64 + d] = f2bf(acc[tm][tn][r] + bb);
      }
    }
  }
}

// ---------------------------------------------------------------------------
// Kernel 2: RoPE on Q and K (bf16, in-place). gridDim.y selects buffer.
// ---------------------------------------------------------------------------
__global__ __launch_bounds__(256) void rope_kernel(bf16_t* __restrict__ Qb,
                                                   bf16_t* __restrict__ Kb)
{
  bf16_t* buf = blockIdx.y ? Kb : Qb;
  int p = blockIdx.x * 256 + threadIdx.x;        // pair index, 4*16*2048*32 total
  int i = p & 31;                                // frequency index
  int t = (p >> 5) & 2047;                       // position
  float invf = __expf(-(float)i * 0.28782313662425575f);  // ln(10000)/32
  float ang = (float)t * invf;
  float sn, cs;
  __sincosf(ang, &sn, &cs);
  size_t idx = (size_t)(p >> 5) * 64 + 2 * i;
  float e = bf2f(buf[idx]), o = bf2f(buf[idx + 1]);
  buf[idx]     = f2bf(e * cs - o * sn);
  buf[idx + 1] = f2bf(e * sn + o * cs);
}

// ---------------------------------------------------------------------------
// Kernel 3: flash attention. One block per (b*h, 64-query tile); 4 waves,
// each wave owns 16 query rows. Online softmax in f32, S and PV via WMMA.
// ---------------------------------------------------------------------------
__global__ __launch_bounds__(128) void attn_kernel(
    const bf16_t* __restrict__ Q, const bf16_t* __restrict__ Kb,
    const bf16_t* __restrict__ Vb, bf16_t* __restrict__ Y)
{
  const int bh = blockIdx.x;                 // 0..63
  const int qt = blockIdx.y;                 // 0..127
  const int b = bh >> 4, h = bh & 15;
  const int tid = threadIdx.x, lane = tid & 31, wave = tid >> 5;
  const int q0 = qt * 64 + wave * 16;

  __shared__ __align__(16) bf16_t Ks[64][72];       // [key][dh]  -> B frags for S
  __shared__ __align__(16) bf16_t Vt[64][72];       // [dh][key]  -> B frags for PV
  __shared__ __align__(16) bf16_t Ps[4][16][72];    // per-wave P  [m][key]

  // Load Q A-fragments straight from global ([B,H,T,Dh] rows are k-contiguous).
  const bf16_t* qbase = Q + ((size_t)bh * 2048 + q0) * 64;
  v16bf qf[2];
  {
    int row = lane & 15;
    int kb0 = (lane & 16) ? 8 : 0;
    #pragma unroll
    for (int c = 0; c < 2; ++c) {
      const bf16_t* p = qbase + (size_t)row * 64 + c * 32;
      U2 t;
      t.a = *(const uint4*)(p + kb0);
      t.b = *(const uint4*)(p + kb0 + 16);
      qf[c] = __builtin_bit_cast(v16bf, t);
    }
  }

  v8f o[4] = {};
  float mrow[8], lrow[8];
  #pragma unroll
  for (int r = 0; r < 8; ++r) { mrow[r] = -1e30f; lrow[r] = 0.0f; }
  const float scale = 0.125f;                 // 1/sqrt(64)
  const int ncol = lane & 15, mo = (lane & 16) ? 8 : 0;

  for (int j = 0; j <= qt; ++j) {
    __syncthreads();                          // protect Ks/Vt from prior readers
    const bf16_t* kbase = Kb + ((size_t)bh * 2048 + j * 64) * 64;
    const bf16_t* vbase = Vb + ((size_t)bh * 2048 + j * 64) * 64;
    for (int c8 = tid; c8 < 512; c8 += 128) { // 64x64 bf16 per tensor, 16B chunks
      int r = c8 >> 3, cc = (c8 & 7) * 8;
      uint4 kd = *(const uint4*)(kbase + (size_t)r * 64 + cc);
      *(uint4*)&Ks[r][cc] = kd;
      uint4 vd = *(const uint4*)(vbase + (size_t)r * 64 + cc);
      unsigned short e[8];
      __builtin_memcpy(e, &vd, 16);
      #pragma unroll
      for (int i = 0; i < 8; ++i) Vt[cc + i][r] = __builtin_bit_cast(bf16_t, e[i]);
    }
    __syncthreads();

    // S = Q * K^T : 4 key-tiles of 16, Dh=64 -> 2 WMMA each.
    v8f s[4];
    #pragma unroll
    for (int nt = 0; nt < 4; ++nt) {
      v8f z = {};
      z = wmma_bf16(qf[0], load_frag_b(&Ks[nt * 16][0],  72, lane), z);
      z = wmma_bf16(qf[1], load_frag_b(&Ks[nt * 16][32], 72, lane), z);
      s[nt] = z;
    }

    // Causal mask + online softmax (row stats across the 16-lane N groups).
    #pragma unroll
    for (int r = 0; r < 8; ++r) {
      int qrow = q0 + r + mo;
      float sv[4], rmax = -1e30f;
      #pragma unroll
      for (int nt = 0; nt < 4; ++nt) {
        int kcol = j * 64 + nt * 16 + ncol;
        float v = s[nt][r] * scale;
        if (kcol > qrow) v = -1e30f;
        sv[nt] = v;
        rmax = fmaxf(rmax, v);
      }
      #pragma unroll
      for (int msk = 1; msk < 16; msk <<= 1) rmax = fmaxf(rmax, __shfl_xor(rmax, msk));
      float mnew  = fmaxf(mrow[r], rmax);
      float alpha = __expf(mrow[r] - mnew);
      float psum  = 0.0f;
      #pragma unroll
      for (int nt = 0; nt < 4; ++nt) {
        float pv = __expf(sv[nt] - mnew);
        sv[nt] = pv;
        psum += pv;
      }
      #pragma unroll
      for (int msk = 1; msk < 16; msk <<= 1) psum += __shfl_xor(psum, msk);
      lrow[r] = lrow[r] * alpha + psum;
      mrow[r] = mnew;
      #pragma unroll
      for (int nt = 0; nt < 4; ++nt) {
        o[nt][r] *= alpha;
        Ps[wave][r + mo][nt * 16 + ncol] = f2bf(sv[nt]);   // C-layout -> LDS
      }
    }
    __syncthreads();                          // publish P before A-layout reload

    // O += P * V : 4 Dh-tiles of 16, 64 keys -> 2 WMMA each.
    #pragma unroll
    for (int nt = 0; nt < 4; ++nt) {
      #pragma unroll
      for (int kc = 0; kc < 2; ++kc) {
        v16bf pa = load_frag_a(&Ps[wave][0][kc * 32], 72, lane);
        v16bf vb = load_frag_b(&Vt[nt * 16][kc * 32], 72, lane);
        o[nt] = wmma_bf16(pa, vb, o[nt]);
      }
    }
  }

  // Normalize and store y as bf16 in [B, T, C] (ready for output projection).
  float inv[8];
  #pragma unroll
  for (int r = 0; r < 8; ++r) inv[r] = 1.0f / lrow[r];
  #pragma unroll
  for (int nt = 0; nt < 4; ++nt) {
    #pragma unroll
    for (int r = 0; r < 8; ++r) {
      int qrow = q0 + r + mo;
      int col  = h * 64 + nt * 16 + ncol;
      Y[((size_t)b * 2048 + qrow) * 1024 + col] = f2bf(o[nt][r] * inv[r]);
    }
  }
}

// ---------------------------------------------------------------------------
// Kernel 4: output projection. y(bf16) x Wo + bo -> f32 d_out.
// ---------------------------------------------------------------------------
__global__ __launch_bounds__(256) void proj_gemm_kernel(
    const bf16_t* __restrict__ Yb, const float* __restrict__ Wo,
    const float* __restrict__ bo, float* __restrict__ Out)
{
  const int N = 1024, K = 1024;
  __shared__ __align__(16) bf16_t As[128][GM_AS];
  __shared__ __align__(16) bf16_t Bt[64][GM_BS];

  const int tid  = threadIdx.x;
  const int lane = tid & 31, wave = tid >> 5;
  const int wm = wave & 3, wn = wave >> 2;
  const int m0 = blockIdx.y * 128, n0 = blockIdx.x * 64;
  const int rowBase = wm * 32, colBase = wn * 32;

  v8f acc[2][2] = {};

  for (int k0 = 0; k0 < K; k0 += 32) {
    __syncthreads();
    #pragma unroll
    for (int i = 0; i < 16; ++i) {
      int idx = tid + i * 256;
      int r = idx >> 5, c = idx & 31;
      As[r][c] = Yb[(size_t)(m0 + r) * K + (k0 + c)];
    }
    #pragma unroll
    for (int i = 0; i < 8; ++i) {
      int idx = tid + i * 256;
      int kk = idx >> 6, n = idx & 63;
      Bt[n][kk] = f2bf(Wo[(size_t)(k0 + kk) * N + (n0 + n)]);
    }
    __syncthreads();

    v16bf a0 = load_frag_a(&As[rowBase][0],      GM_AS, lane);
    v16bf a1 = load_frag_a(&As[rowBase + 16][0], GM_AS, lane);
    v16bf b0 = load_frag_b(&Bt[colBase][0],      GM_BS, lane);
    v16bf b1 = load_frag_b(&Bt[colBase + 16][0], GM_BS, lane);
    acc[0][0] = wmma_bf16(a0, b0, acc[0][0]);
    acc[0][1] = wmma_bf16(a0, b1, acc[0][1]);
    acc[1][0] = wmma_bf16(a1, b0, acc[1][0]);
    acc[1][1] = wmma_bf16(a1, b1, acc[1][1]);
  }

  const int ncol = lane & 15, mo = (lane & 16) ? 8 : 0;
  #pragma unroll
  for (int tm = 0; tm < 2; ++tm) {
    #pragma unroll
    for (int tn = 0; tn < 2; ++tn) {
      int n = n0 + colBase + tn * 16 + ncol;
      float bb = bo[n];
      #pragma unroll
      for (int r = 0; r < 8; ++r) {
        int m = m0 + rowBase + tm * 16 + r + mo;
        Out[(size_t)m * N + n] = acc[tm][tn][r] + bb;
      }
    }
  }
}

// ---------------------------------------------------------------------------
// Host launcher.
// Inputs: x, Wq, bq, Wk, bk, Wv, bv, Wo, bo (all f32). Output: f32 [4,2048,1024].
// Workspace: Q,K,V,Y bf16 buffers (16 MB each -> 64 MB total).
// ---------------------------------------------------------------------------
extern "C" void kernel_launch(void* const* d_in, const int* in_sizes, int n_in,
                              void* d_out, int out_size, void* d_ws, size_t ws_size,
                              hipStream_t stream) {
  (void)in_sizes; (void)n_in; (void)out_size; (void)ws_size;
  const float* x  = (const float*)d_in[0];
  const float* Wq = (const float*)d_in[1];
  const float* bq = (const float*)d_in[2];
  const float* Wk = (const float*)d_in[3];
  const float* bk = (const float*)d_in[4];
  const float* Wv = (const float*)d_in[5];
  const float* bv = (const float*)d_in[6];
  const float* Wo = (const float*)d_in[7];
  const float* bo = (const float*)d_in[8];
  float* out = (float*)d_out;

  char* ws = (char*)d_ws;
  const size_t SZ = (size_t)8192 * 1024 * sizeof(unsigned short);  // 16 MB
  bf16_t* q = (bf16_t*)(ws);
  bf16_t* k = (bf16_t*)(ws + SZ);
  bf16_t* v = (bf16_t*)(ws + 2 * SZ);
  bf16_t* y = (bf16_t*)(ws + 3 * SZ);

  // 1) QKV projections (z = 0,1,2 -> Q,K,V)
  qkv_gemm_kernel<<<dim3(16, 64, 3), 256, 0, stream>>>(x, Wq, Wk, Wv, bq, bk, bv,
                                                       q, k, v);
  // 2) RoPE on Q and K (4*16*2048*32 pairs / 256)
  rope_kernel<<<dim3(16384, 2), 256, 0, stream>>>(q, k);
  // 3) Flash attention: (B*H = 64) x (T/64 = 128) tiles, 128 threads
  attn_kernel<<<dim3(64, 128), 128, 0, stream>>>(q, k, v, y);
  // 4) Output projection -> f32 out
  proj_gemm_kernel<<<dim3(16, 64), 256, 0, stream>>>(y, Wo, bo, out);
}
